// G2_87857851007234
// MI455X (gfx1250) — matrix operand order repaired and checked by reading, not compiled, not run on hardware
//
#include <hip/hip_runtime.h>
#include <hip/hip_bf16.h>

typedef __attribute__((ext_vector_type(16))) _Float16 v16h;
typedef __attribute__((ext_vector_type(8)))  float    v8f;

#define NHID 64

// ---------------------------------------------------------------------------
// helpers
// ---------------------------------------------------------------------------
__device__ __forceinline__ void splitf(float x, _Float16& h, _Float16& l) {
  _Float16 hh = (_Float16)x;
  h = hh;
  l = (_Float16)(x - (float)hh);
}

// A fragment (16x32, f16): lane (0..31): row = row0 + (lane&15);
// K = k0 + ((lane&16)?8:0) + (e<8 ? e : e+8)   [ISA 16-bit A 16x32 layout]
__device__ __forceinline__ void load_a(const float* __restrict__ M, int ld,
                                       int row0, int k0, float scale, int lane,
                                       v16h& hi, v16h& lo) {
  int r  = row0 + (lane & 15);
  int kb = k0 + ((lane & 16) ? 8 : 0);
  const float* p = M + (size_t)r * ld + kb;
#pragma unroll
  for (int e = 0; e < 16; ++e) {
    int k = (e < 8) ? e : (e + 8);
    float x = p[k] * scale;
    _Float16 h, l; splitf(x, h, l);
    hi[e] = h; lo[e] = l;
  }
}

// B fragment (32x16, f16) from row-major W[ld columns]:
// lane: col = col0 + (lane&15); K = k0 + ((lane&16)?8:0) + (e<8 ? e : e+8)   [ISA B layout]
__device__ __forceinline__ void load_b(const float* __restrict__ W, int ld,
                                       int k0, int col0, int lane,
                                       v16h& hi, v16h& lo) {
  int cc = col0 + (lane & 15);
  int kb = k0 + ((lane & 16) ? 8 : 0);
  const float* p = W + (size_t)kb * ld + cc;
#pragma unroll
  for (int e = 0; e < 16; ++e) {
    int k = (e < 8) ? e : (e + 8);
    float x = p[(size_t)k * ld];
    _Float16 h, l; splitf(x, h, l);
    hi[e] = h; lo[e] = l;
  }
}

// split-f16 3-product accumulate: C += A*B in ~fp32 precision
__device__ __forceinline__ v8f wmma3(const v16h& ah, const v16h& al,
                                     const v16h& bh, const v16h& bl, v8f c) {
  c = __builtin_amdgcn_wmma_f32_16x16x32_f16(false, ah, false, bh, (short)0, c, false, false);
  c = __builtin_amdgcn_wmma_f32_16x16x32_f16(false, al, false, bh, (short)0, c, false, false);
  c = __builtin_amdgcn_wmma_f32_16x16x32_f16(false, ah, false, bl, (short)0, c, false, false);
  return c;
}

// ---------------------------------------------------------------------------
// kernels
// ---------------------------------------------------------------------------
__global__ void zero_f32(float* __restrict__ p, long long n) {
  long long i = (long long)blockIdx.x * blockDim.x + threadIdx.x;
  long long stride = (long long)gridDim.x * blockDim.x;
  for (; i < n; i += stride) p[i] = 0.0f;
}

// phase 1: acc[dst] += X[src]; cntd[dst] += 1   (wave32 per edge, float2/lane)
__global__ void edge_scatter1(const float* __restrict__ X,
                              const int* __restrict__ ei,
                              float* __restrict__ acc,
                              float* __restrict__ cntd, int E) {
  long long gid = (long long)blockIdx.x * blockDim.x + threadIdx.x;
  int e    = (int)(gid >> 5);
  int lane = (int)(gid & 31);
  if (e >= E) return;
  int s = ei[e];
  int d = ei[(size_t)E + e];
  const float2 v = *(const float2*)(X + (size_t)s * NHID + lane * 2);
  float* ad = acc + (size_t)d * NHID + lane * 2;
  atomicAdd(ad,     v.x);
  atomicAdd(ad + 1, v.y);
  if (lane == 0) atomicAdd(cntd + d, 1.0f);
}

// H = relu(X@Ws + (acc/max(cntd,1))@Wn + bias)   — 48 v_wmma per wave
__global__ void sage_gemm_kernel(const float* __restrict__ X,
                                 const float* __restrict__ acc,
                                 const float* __restrict__ cntd,
                                 const float* __restrict__ Ws,
                                 const float* __restrict__ Wn,
                                 const float* __restrict__ bias,
                                 float* __restrict__ H, int nTiles) {
  int wave = (int)(((long long)blockIdx.x * blockDim.x + threadIdx.x) >> 5);
  int lane = threadIdx.x & 31;
  if (wave >= nTiles) return;              // wave-uniform: EXEC stays all-1s
  int row0 = wave * 16;
  int r = row0 + (lane & 15);
  float inv = 1.0f / fmaxf(cntd[r], 1.0f);

  v16h axh[2], axl[2], amh[2], aml[2];
#pragma unroll
  for (int kt = 0; kt < 2; ++kt) {
    load_a(X,   NHID, row0, kt * 32, 1.0f, lane, axh[kt], axl[kt]);
    load_a(acc, NHID, row0, kt * 32, inv,  lane, amh[kt], aml[kt]);
  }
  int col   = lane & 15;
  int rbase = (lane & 16) ? 8 : 0;
#pragma unroll
  for (int nt = 0; nt < 4; ++nt) {
    v8f c = {};
#pragma unroll
    for (int kt = 0; kt < 2; ++kt) {
      v16h bh, bl;
      load_b(Ws, NHID, kt * 32, nt * 16, lane, bh, bl);
      c = wmma3(axh[kt], axl[kt], bh, bl, c);
      load_b(Wn, NHID, kt * 32, nt * 16, lane, bh, bl);
      c = wmma3(amh[kt], aml[kt], bh, bl, c);
    }
    int gc = nt * 16 + col;
    float bv = bias[gc];
#pragma unroll
    for (int v = 0; v < 8; ++v) {
      float val = fmaxf(c[v] + bv, 0.0f);
      H[(size_t)(row0 + rbase + v) * NHID + gc] = val;
    }
  }
}

// Ap = H @ Q_w[0:64,:],  Bp = H @ Q_w[64:128,:]   — 48 v_wmma per wave
__global__ void q_gemm_kernel(const float* __restrict__ H,
                              const float* __restrict__ Qw,
                              float* __restrict__ Ap,
                              float* __restrict__ Bp, int nTiles) {
  int wave = (int)(((long long)blockIdx.x * blockDim.x + threadIdx.x) >> 5);
  int lane = threadIdx.x & 31;
  if (wave >= nTiles) return;
  int row0 = wave * 16;

  v16h ahh[2], ahl[2];
#pragma unroll
  for (int kt = 0; kt < 2; ++kt)
    load_a(H, NHID, row0, kt * 32, 1.0f, lane, ahh[kt], ahl[kt]);

  const float* Qt  = Qw;                 // rows 0..63
  const float* Qb2 = Qw + 64 * NHID;     // rows 64..127
  int col   = lane & 15;
  int rbase = (lane & 16) ? 8 : 0;
#pragma unroll
  for (int nt = 0; nt < 4; ++nt) {
    v8f ca = {}, cb = {};
#pragma unroll
    for (int kt = 0; kt < 2; ++kt) {
      v16h bh, bl;
      load_b(Qt, NHID, kt * 32, nt * 16, lane, bh, bl);
      ca = wmma3(ahh[kt], ahl[kt], bh, bl, ca);
      load_b(Qb2, NHID, kt * 32, nt * 16, lane, bh, bl);
      cb = wmma3(ahh[kt], ahl[kt], bh, bl, cb);
    }
    int gc = nt * 16 + col;
#pragma unroll
    for (int v = 0; v < 8; ++v) {
      size_t idx = (size_t)(row0 + rbase + v) * NHID + gc;
      Ap[idx] = ca[v];
      Bp[idx] = cb[v];
    }
  }
}

// phase 2: S1[src] += Bp[dst]; S2[src] += Bp[dst]^2; cnts[src] += 1
__global__ void edge_scatter2(const float* __restrict__ Bp,
                              const int* __restrict__ ei,
                              float* __restrict__ S1,
                              float* __restrict__ S2,
                              float* __restrict__ cnts, int E) {
  long long gid = (long long)blockIdx.x * blockDim.x + threadIdx.x;
  int e    = (int)(gid >> 5);
  int lane = (int)(gid & 31);
  if (e >= E) return;
  int s = ei[e];
  int d = ei[(size_t)E + e];
  const float2 b = *(const float2*)(Bp + (size_t)d * NHID + lane * 2);
  float* p1 = S1 + (size_t)s * NHID + lane * 2;
  float* p2 = S2 + (size_t)s * NHID + lane * 2;
  atomicAdd(p1,     b.x);
  atomicAdd(p1 + 1, b.y);
  atomicAdd(p2,     b.x * b.x);
  atomicAdd(p2 + 1, b.y * b.y);
  if (lane == 0) atomicAdd(cnts + s, 1.0f);
}

// out = tanh( a^2 + (2a*S1 + S2)/cnt ),  a = Ap + Q_b;  empty segment -> 0
__global__ void finalize_kernel(const float* __restrict__ Ap,
                                const float* __restrict__ S1,
                                const float* __restrict__ S2,
                                const float* __restrict__ cnts,
                                const float* __restrict__ Qb,
                                float* __restrict__ out, int N) {
  long long gid = (long long)blockIdx.x * blockDim.x + threadIdx.x;
  if (gid >= (long long)N * NHID) return;
  int node = (int)(gid >> 6);
  int c    = (int)(gid & 63);
  float cnt = cnts[node];
  float o = 0.0f;
  if (cnt > 0.5f) {
    float a = Ap[gid] + Qb[c];
    float m = fmaf(a, a, fmaf(2.0f * a, S1[gid], S2[gid]) / cnt);
    o = tanhf(m);
  }
  out[gid] = o;
}

// ---------------------------------------------------------------------------
// launch
// ---------------------------------------------------------------------------
extern "C" void kernel_launch(void* const* d_in, const int* in_sizes, int n_in,
                              void* d_out, int out_size, void* d_ws, size_t ws_size,
                              hipStream_t stream) {
  const float* X  = (const float*)d_in[0];
  const int*   ei = (const int*)d_in[1];
  const float* Ws = (const float*)d_in[2];
  const float* Wn = (const float*)d_in[3];
  const float* bs = (const float*)d_in[4];
  const float* Qw = (const float*)d_in[5];
  const float* Qb = (const float*)d_in[6];
  int N = in_sizes[0] / NHID;
  int E = in_sizes[1] / 2;

  // workspace layout (floats); acc reused as S1, H reused as S2
  float* acc  = (float*)d_ws;                   // N*64  (later S1)
  float* H    = acc  + (size_t)N * NHID;        // N*64  (later S2)
  float* Ap   = H    + (size_t)N * NHID;        // N*64
  float* Bp   = Ap   + (size_t)N * NHID;        // N*64
  float* cntd = Bp   + (size_t)N * NHID;        // N
  float* cnts = cntd + (size_t)N;               // N

  long long edgeThreads = (long long)E * 32;
  int edgeBlocks = (int)((edgeThreads + 255) / 256);
  int nTiles = N / 16;                          // N=50000 -> 3125 exact
  int gemmBlocks = (nTiles + 7) / 8;            // 8 waves / 256-thread block

  zero_f32<<<2048, 256, 0, stream>>>(acc, (long long)N * NHID);
  zero_f32<<<256, 256, 0, stream>>>(cntd, (long long)N);

  edge_scatter1<<<edgeBlocks, 256, 0, stream>>>(X, ei, acc, cntd, E);

  sage_gemm_kernel<<<gemmBlocks, 256, 0, stream>>>(X, acc, cntd, Ws, Wn, bs, H, nTiles);
  q_gemm_kernel<<<gemmBlocks, 256, 0, stream>>>(H, Qw, Ap, Bp, nTiles);

  zero_f32<<<2048, 256, 0, stream>>>(acc, (long long)N * NHID * 2);  // S1+S2 contiguous
  zero_f32<<<256, 256, 0, stream>>>(cnts, (long long)N);

  edge_scatter2<<<edgeBlocks, 256, 0, stream>>>(Bp, ei, acc, H, cnts, E);

  long long outElems = (long long)N * NHID;
  finalize_kernel<<<(int)((outElems + 255) / 256), 256, 0, stream>>>(
      Ap, acc, H, cnts, Qb, (float*)d_out, N);
}
